// CRFillModule_56727928046373
// MI455X (gfx1250) — matrix operand order repaired and checked by reading, not compiled, not run on hardware
//
#include <hip/hip_runtime.h>
#include <hip/hip_bf16.h>

// ---------------- problem constants (from reference) ----------------
#define BB     2
#define CC     192
#define HH     48
#define WW     48
#define PS     3
#define WO     46            // Ho = Wo = 46
#define NREAL  (WO * WO)     // 2116 patches
#define NPAD   2176          // 17 * 128 (and 34 * 64)
#define DDIM   (CC * PS * PS)// 1728 = 27*64
#define RECON_ELEMS ((size_t)BB * CC * HH * WW)   // 884736
#define SCALE  10.0f
#define LOSS_PART_BLOCKS 512

typedef _Float16 h16 __attribute__((ext_vector_type(16)));
typedef _Float16 h8  __attribute__((ext_vector_type(8)));
typedef float    v8f __attribute__((ext_vector_type(8)));

// ---------------- prep: patches -> normalized f16 Q, P, Pt ----------------
__global__ __launch_bounds__(256) void prep_kernel(
    const float* __restrict__ feat, const float* __restrict__ mask,
    _Float16* __restrict__ Q, _Float16* __restrict__ P, _Float16* __restrict__ Pt)
{
    int tid = blockIdx.x * 256 + threadIdx.x;           // (b, c, n) with n fastest
    int n  = tid % NPAD;
    int bc = tid / NPAD;
    int c  = bc % CC;
    int b  = bc / CC;
    if (b >= BB) return;

    _Float16* qrow = Q + ((size_t)b * NPAD + n) * DDIM + c * 9;
    _Float16* prow = P + ((size_t)b * NPAD + n) * DDIM + c * 9;

    if (n >= NREAL) {  // zero padding rows / columns
        #pragma unroll
        for (int k = 0; k < 9; k++) {
            qrow[k] = (_Float16)0.0f;
            prow[k] = (_Float16)0.0f;
            Pt[((size_t)b * DDIM + c * 9 + k) * NPAD + n] = (_Float16)0.0f;
        }
        return;
    }

    int y = n / WO, x = n % WO;
    float pv[9], qv[9];
    float sp = 0.0f, sq = 0.0f;
    #pragma unroll
    for (int i = 0; i < PS; i++) {
        #pragma unroll
        for (int j = 0; j < PS; j++) {
            int k = i * PS + j;
            float f = feat[(((size_t)b * CC + c) * HH + (y + i)) * WW + (x + j)];
            float m = mask[((size_t)b * HH + (y + i)) * WW + (x + j)];
            pv[k] = f;                sp += f * f;
            float fq = f * (1.0f - m);
            qv[k] = fq;               sq += fq * fq;
        }
    }
    float rp = 1.0f / fmaxf(sqrtf(sp), 1e-12f);
    float rq = 1.0f / fmaxf(sqrtf(sq), 1e-12f);
    #pragma unroll
    for (int k = 0; k < 9; k++) {
        _Float16 ph = (_Float16)(pv[k] * rp);
        prow[k] = ph;
        qrow[k] = (_Float16)(qv[k] * rq);
        Pt[((size_t)b * DDIM + c * 9 + k) * NPAD + n] = ph;
    }
}

// ---------------- GEMM1: sim = SCALE * Q @ P^T  (f16 -> f32) ----------------
// WG tile 128(n) x 128(m); 8 waves as 4(n) x 2(m); wave tile 32x64 = 2x4 WMMAs.
// Double-buffered k-loop, unrolled by 64 (DDIM % 64 == 0).
__global__ __launch_bounds__(256) void gemm1_sim(
    const _Float16* __restrict__ Q, const _Float16* __restrict__ P,
    float* __restrict__ sim)
{
    const int b    = blockIdx.z;
    const int n0   = blockIdx.y * 128;
    const int m0   = blockIdx.x * 128;
    const int w    = threadIdx.x >> 5;
    const int lane = threadIdx.x & 31;
    const int wr   = w >> 1;          // 0..3 -> n offset 32*wr
    const int wc   = w & 1;           // 0..1 -> m offset 64*wc
    const int lr   = lane & 15;
    const int lh   = lane >> 4;       // lane half

    const _Float16* Qb = Q + (size_t)b * NPAD * DDIM;
    const _Float16* Pb = P + (size_t)b * NPAD * DDIM;
    float* simb = sim + (size_t)b * NPAD * NPAD;

    const _Float16* qrow[2];
    const _Float16* prow[4];
    #pragma unroll
    for (int sn = 0; sn < 2; sn++)
        qrow[sn] = Qb + (size_t)(n0 + wr * 32 + sn * 16 + lr) * DDIM;
    #pragma unroll
    for (int sm = 0; sm < 4; sm++)
        prow[sm] = Pb + (size_t)(m0 + wc * 64 + sm * 16 + lr) * DDIM;

    v8f zero = {};
    v8f acc[2][4];
    #pragma unroll
    for (int i = 0; i < 2; i++)
        #pragma unroll
        for (int j = 0; j < 4; j++) acc[i][j] = zero;

    h16 a0[2], a1[2], b0[4], b1[4];

#define LOAD_G1(AV, BV, KK)                                          \
    _Pragma("unroll") for (int sn = 0; sn < 2; sn++) {               \
        union { h16 v; h8 p[2]; } u;                                 \
        u.p[0] = *(const h8*)(qrow[sn] + (KK) + lh * 16);            \
        u.p[1] = *(const h8*)(qrow[sn] + (KK) + lh * 16 + 8);        \
        AV[sn] = u.v; }                                              \
    _Pragma("unroll") for (int sm = 0; sm < 4; sm++)                 \
        BV[sm] = *(const h16*)(prow[sm] + (KK) + lh * 16);

#define WMMA_G1(AV, BV)                                              \
    _Pragma("unroll") for (int sn = 0; sn < 2; sn++)                 \
    _Pragma("unroll") for (int sm = 0; sm < 4; sm++)                 \
        acc[sn][sm] = __builtin_amdgcn_wmma_f32_16x16x32_f16(        \
            false, AV[sn], false, BV[sm], (short)0, acc[sn][sm], false, false);

    LOAD_G1(a0, b0, 0)
    for (int kk = 0; kk < DDIM; kk += 64) {
        LOAD_G1(a1, b1, kk + 32)
        WMMA_G1(a0, b0)
        if (kk + 64 < DDIM) { LOAD_G1(a0, b0, kk + 64) }
        WMMA_G1(a1, b1)
    }
#undef LOAD_G1
#undef WMMA_G1

    #pragma unroll
    for (int sn = 0; sn < 2; sn++)
        #pragma unroll
        for (int sm = 0; sm < 4; sm++) {
            int col   = m0 + wc * 64 + sm * 16 + lr;
            int rbase = n0 + wr * 32 + sn * 16 + lh * 8;
            #pragma unroll
            for (int i = 0; i < 8; i++)
                simb[(size_t)(rbase + i) * NPAD + col] = acc[sn][sm][i] * SCALE;
        }
}

// ---------------- row softmax: sim (f32) -> attn (f16), pads zeroed ----------------
__global__ __launch_bounds__(256) void softmax_rows(
    const float* __restrict__ sim, _Float16* __restrict__ attn)
{
    const int n = blockIdx.x;
    const int b = blockIdx.y;
    const int t = threadIdx.x;
    const float* row = sim + ((size_t)b * NPAD + n) * NPAD;
    _Float16* arow = attn + ((size_t)b * NPAD + n) * NPAD;
    __shared__ float red[256];

    if (n >= NREAL) {   // padding rows: write zeros (safe operand for GEMM2)
        for (int m = t; m < NPAD; m += 256) arow[m] = (_Float16)0.0f;
        return;
    }

    float mx = -3.4e38f;
    for (int m = t; m < NREAL; m += 256) mx = fmaxf(mx, row[m]);
    red[t] = mx; __syncthreads();
    for (int s = 128; s > 0; s >>= 1) { if (t < s) red[t] = fmaxf(red[t], red[t + s]); __syncthreads(); }
    mx = red[0]; __syncthreads();

    float sum = 0.0f;
    for (int m = t; m < NREAL; m += 256) sum += __expf(row[m] - mx);
    red[t] = sum; __syncthreads();
    for (int s = 128; s > 0; s >>= 1) { if (t < s) red[t] += red[t + s]; __syncthreads(); }
    float inv = 1.0f / red[0];

    for (int m = t; m < NREAL; m += 256) arow[m] = (_Float16)(__expf(row[m] - mx) * inv);
    for (int m = NREAL + t; m < NPAD; m += 256) arow[m] = (_Float16)0.0f;
}

// ---------------- GEMM2: recon_p^T = Pt @ attn^T ----------------
// Output tile rows = d (Pt rows), cols = n (attn rows). A = Pt, B = attn rows
// (both contiguous fragment loads). WG tile 64(d) x 128(n); 4 waves 2(d) x 2(n);
// wave tile 32x64 = 2x4 WMMAs. Double-buffered m-loop (NPAD % 64 == 0).
__global__ __launch_bounds__(128) void gemm2_recon(
    const _Float16* __restrict__ Pt, const _Float16* __restrict__ attn,
    float* __restrict__ outrp)
{
    const int b    = blockIdx.z;
    const int d0   = blockIdx.x * 64;
    const int n0   = blockIdx.y * 128;
    const int w    = threadIdx.x >> 5;  // 0..3
    const int lane = threadIdx.x & 31;
    const int wr   = w >> 1;            // 0..1 -> d offset 32*wr
    const int wc   = w & 1;             // 0..1 -> n offset 64*wc
    const int lr   = lane & 15;
    const int lh   = lane >> 4;

    const _Float16* Ptb = Pt   + (size_t)b * DDIM * NPAD;
    const _Float16* Ab  = attn + (size_t)b * NPAD * NPAD;

    const _Float16* arow[2];
    const _Float16* brow[4];
    #pragma unroll
    for (int sd = 0; sd < 2; sd++)
        arow[sd] = Ptb + (size_t)(d0 + wr * 32 + sd * 16 + lr) * NPAD;
    #pragma unroll
    for (int sn = 0; sn < 4; sn++)
        brow[sn] = Ab + (size_t)(n0 + wc * 64 + sn * 16 + lr) * NPAD;

    v8f zero = {};
    v8f acc[2][4];
    #pragma unroll
    for (int i = 0; i < 2; i++)
        #pragma unroll
        for (int j = 0; j < 4; j++) acc[i][j] = zero;

    h16 a0[2], a1[2], b0[4], b1[4];

#define LOAD_G2(AV, BV, MM)                                          \
    _Pragma("unroll") for (int sd = 0; sd < 2; sd++) {               \
        union { h16 v; h8 p[2]; } u;                                 \
        u.p[0] = *(const h8*)(arow[sd] + (MM) + lh * 16);            \
        u.p[1] = *(const h8*)(arow[sd] + (MM) + lh * 16 + 8);        \
        AV[sd] = u.v; }                                              \
    _Pragma("unroll") for (int sn = 0; sn < 4; sn++)                 \
        BV[sn] = *(const h16*)(brow[sn] + (MM) + lh * 16);

#define WMMA_G2(AV, BV)                                              \
    _Pragma("unroll") for (int sd = 0; sd < 2; sd++)                 \
    _Pragma("unroll") for (int sn = 0; sn < 4; sn++)                 \
        acc[sd][sn] = __builtin_amdgcn_wmma_f32_16x16x32_f16(        \
            false, AV[sd], false, BV[sn], (short)0, acc[sd][sn], false, false);

    LOAD_G2(a0, b0, 0)
    for (int mm = 0; mm < NPAD; mm += 64) {
        LOAD_G2(a1, b1, mm + 32)
        WMMA_G2(a0, b0)
        if (mm + 64 < NPAD) { LOAD_G2(a0, b0, mm + 64) }
        WMMA_G2(a1, b1)
    }
#undef LOAD_G2
#undef WMMA_G2

    // scatter to recon_p [B, C, N, 9]; d = c*9 + k
    #pragma unroll
    for (int sd = 0; sd < 2; sd++)
        #pragma unroll
        for (int sn = 0; sn < 4; sn++) {
            int n = n0 + wc * 64 + sn * 16 + lr;
            if (n < NREAL) {
                #pragma unroll
                for (int i = 0; i < 8; i++) {
                    int d = d0 + wr * 32 + sd * 16 + lh * 8 + i;
                    int c = d / 9, k = d % 9;
                    outrp[(((size_t)b * CC + c) * NREAL + n) * 9 + k] = acc[sd][sn][i];
                }
            }
        }
}

// ---------------- recon = 0 ----------------
__global__ __launch_bounds__(256) void zero_recon(float4* __restrict__ out)
{
    size_t i = (size_t)blockIdx.x * 256 + threadIdx.x;
    float4 z; z.x = 0.f; z.y = 0.f; z.z = 0.f; z.w = 0.f;
    out[i] = z;
}

// ---------------- cr_loss = mean((feat*(1-mask))^2), deterministic two-stage ----------------
__global__ __launch_bounds__(256) void loss_partial(
    const float* __restrict__ feat, const float* __restrict__ mask,
    float* __restrict__ part)
{
    __shared__ float red[256];
    int t = threadIdx.x;
    float s = 0.0f;
    const size_t total = RECON_ELEMS;
    for (size_t i = (size_t)blockIdx.x * 256 + t; i < total; i += (size_t)gridDim.x * 256) {
        size_t hw = i % (HH * WW);
        size_t bc = i / (HH * WW);
        size_t b  = bc / CC;
        float m = mask[b * (HH * WW) + hw];
        float d = feat[i] * (1.0f - m);
        s += d * d;
    }
    red[t] = s; __syncthreads();
    for (int st = 128; st > 0; st >>= 1) { if (t < st) red[t] += red[t + st]; __syncthreads(); }
    if (t == 0) part[blockIdx.x] = red[0];
}

__global__ __launch_bounds__(256) void loss_final(
    const float* __restrict__ part, float* __restrict__ out_loss)
{
    __shared__ float red[256];
    int t = threadIdx.x;
    float s = part[t] + part[t + 256];   // LOSS_PART_BLOCKS == 512
    red[t] = s; __syncthreads();
    for (int st = 128; st > 0; st >>= 1) { if (t < st) red[t] += red[t + st]; __syncthreads(); }
    if (t == 0) *out_loss = red[0] / (float)RECON_ELEMS;
}

// ---------------- launch ----------------
extern "C" void kernel_launch(void* const* d_in, const int* in_sizes, int n_in,
                              void* d_out, int out_size, void* d_ws, size_t ws_size,
                              hipStream_t stream)
{
    const float* feat = (const float*)d_in[0];
    const float* mask = (const float*)d_in[1];
    float* out = (float*)d_out;

    // workspace layout (all regions 256B aligned)
    char* ws = (char*)d_ws;
    const size_t szQ    = (size_t)BB * NPAD * DDIM * sizeof(_Float16);   // 15,040,512
    const size_t szSim  = (size_t)BB * NPAD * NPAD * sizeof(float);      // 37,879,808
    const size_t szAttn = (size_t)BB * NPAD * NPAD * sizeof(_Float16);   // 18,939,904
    _Float16* Q    = (_Float16*)(ws);
    _Float16* P    = (_Float16*)(ws + szQ);
    _Float16* Pt   = (_Float16*)(ws + 2 * szQ);
    float*    sim  = (float*)   (ws + 3 * szQ);
    _Float16* attn = (_Float16*)(ws + 3 * szQ + szSim);
    float*    part = (float*)   (ws + 3 * szQ + szSim + szAttn);

    float* out_recon = out;                          // [B,C,H,W] zeros
    float* out_loss  = out + RECON_ELEMS;            // scalar
    float* out_rp    = out + RECON_ELEMS + 1;        // [B,C,N,9]

    // 1) trivial outputs
    zero_recon<<<dim3((unsigned)(RECON_ELEMS / 4 / 256)), 256, 0, stream>>>((float4*)out_recon);
    loss_partial<<<dim3(LOSS_PART_BLOCKS), 256, 0, stream>>>(feat, mask, part);
    loss_final<<<dim3(1), 256, 0, stream>>>(part, out_loss);

    // 2) normalized patch matrices (padded)
    {
        unsigned total = (unsigned)((size_t)BB * CC * NPAD);
        prep_kernel<<<dim3(total / 256), 256, 0, stream>>>(feat, mask, Q, P, Pt);
    }

    // 3) sim = SCALE * Q @ P^T
    gemm1_sim<<<dim3(NPAD / 128, NPAD / 128, BB), 256, 0, stream>>>(Q, P, sim);

    // 4) softmax rows -> attn (f16)
    softmax_rows<<<dim3(NPAD, BB), 256, 0, stream>>>(sim, attn);

    // 5) recon_p^T = Pt @ attn^T, scattered to [B,C,N,9]
    gemm2_recon<<<dim3(DDIM / 64, NPAD / 128, BB), 128, 0, stream>>>(Pt, attn, out_rp);
}
